// Attention_66511863546284
// MI455X (gfx1250) — matrix operand order repaired and checked by reading, not compiled, not run on hardware
//
#include <hip/hip_runtime.h>
#include <math.h>

// ---------------------------------------------------------------------------
// Problem constants (from reference): B=64, H=256, S=4096, W:[H,3H], v:[H]
// scores[b,s] = softmax_s( sum_h v[h] * tanh( (W1@static + W2@dyn)[h,s] + bias[b,h] ) )
// bias[b,h]  = sum_k W[h, 512+k] * decoder_hidden[b,k]
// ---------------------------------------------------------------------------
#define BATCH 64
#define HDIM  256
#define SDIM  4096
#define WLD   768          // W row stride (3H)
#define NTILE 64           // columns of S per workgroup
#define KSTEP 32           // WMMA K per step
#define SLAB_LD 68         // fp32 row stride in LDS (64 data + 4 pad dwords via TDM)
#define SLAB_BYTES (KSTEP * SLAB_LD * 4)   // 8704 B per slab

typedef __attribute__((ext_vector_type(16))) __bf16 v16bf;
typedef __attribute__((ext_vector_type(8)))  float  v8f;
typedef __attribute__((ext_vector_type(4)))  float  v4f;
typedef __attribute__((ext_vector_type(4)))  unsigned int u32x4;
typedef __attribute__((ext_vector_type(8)))  int    i32x8;
typedef __attribute__((ext_vector_type(4)))  int    i32x4;

// ---------------------------------------------------------------------------
// Kernel A: bias[b,h] = W[h, 512:768] . hidden[b, :]
// ---------------------------------------------------------------------------
__global__ __launch_bounds__(256) void att_bias_kernel(
    const float* __restrict__ W,
    const float* __restrict__ dec_hidden,   // [1,B,H] -> flat [B*H]
    float* __restrict__ bias)               // [B,H]
{
    const int b = blockIdx.x;
    const int h = threadIdx.x;
    __shared__ float hid[HDIM];
    hid[h] = dec_hidden[b * HDIM + h];
    __syncthreads();
    const float* wr = W + (size_t)h * WLD + 2 * HDIM;
    float acc = 0.f;
#pragma unroll 8
    for (int k = 0; k < HDIM; ++k) acc = fmaf(wr[k], hid[k], acc);
    bias[b * HDIM + h] = acc;
}

// ---------------------------------------------------------------------------
// Kernel B: fused GEMM (bf16 WMMA) + tanh + v-dot -> scores[b, s] (pre-softmax)
// Workgroup: 256 threads = 8 waves. blockIdx.x: S tile, blockIdx.y: batch.
// Wave w computes h-rows [32w, 32w+32) for all NTILE columns.
// Energy slabs (32 x 64 fp32) staged into LDS by the Tensor Data Mover,
// double-buffered so the DMA of slab kk+1 overlaps WMMA compute on slab kk.
// ---------------------------------------------------------------------------
struct SharedMem {
    float slab[2][KSTEP][SLAB_LD];   // MUST be first: D#.lds_addr = buf * SLAB_BYTES
    float v[HDIM];
    float bias[HDIM];
    float score[NTILE];
};

__device__ __forceinline__ void tdm_issue_slab(unsigned long long gaddr,
                                               unsigned int lds_addr) {
    u32x4 g0;
    g0[0] = 1u;                                          // count=1, user mode
    g0[1] = lds_addr;                                    // LDS byte offset of slab
    g0[2] = (unsigned int)(gaddr & 0xFFFFFFFFull);       // global_addr[31:0]
    g0[3] = (unsigned int)((gaddr >> 32) & 0x01FFFFFFull)
          | (2u << 30);                                  // global_addr[56:32] | type=2

    i32x8 g1;
    g1[0] = (int)((2u << 16)      // data_size = 4B
                | (1u << 20)      // pad_enable (LDS row padding)
                | (5u << 22)      // pad_interval: 64 DWORDs
                | (3u << 25));    // pad_amount:   4 DWORDs  -> 68-dword LDS rows
    g1[1] = (int)((SDIM & 0xFFFFu) << 16);               // tensor_dim0[15:0]
    g1[2] = (int)(((SDIM >> 16) & 0xFFFFu)               // tensor_dim0[31:16]
                | ((HDIM & 0xFFFFu) << 16));             // tensor_dim1[15:0]
    g1[3] = (int)(((HDIM >> 16) & 0xFFFFu)               // tensor_dim1[31:16]
                | ((unsigned)NTILE << 16));              // tile_dim0 = 64
    g1[4] = (int)((unsigned)KSTEP);                      // tile_dim1 = 32, tile_dim2 = 0
    g1[5] = (int)SDIM;                                   // tensor_dim0_stride[31:0]
    g1[6] = 0;                                           // stride[47:32] | dim1_stride[15:0]
    g1[7] = 0;

    i32x4 g2 = {0, 0, 0, 0};
    i32x4 g3 = {0, 0, 0, 0};

#if defined(__clang_major__) && (__clang_major__ >= 23)
    i32x8 g4 = {0, 0, 0, 0, 0, 0, 0, 0};
    __builtin_amdgcn_tensor_load_to_lds(g0, g1, g2, g3, g4, 0);
#else
    __builtin_amdgcn_tensor_load_to_lds(g0, g1, g2, g3, 0);
#endif
}

__global__ __launch_bounds__(256) void att_gemm_kernel(
    const float* __restrict__ s_enc,   // [B,H,S]
    const float* __restrict__ d_enc,   // [B,H,S]
    const float* __restrict__ W,       // [H,3H]
    const float* __restrict__ vvec,    // [H]
    const float* __restrict__ bias,    // [B,H]
    float* __restrict__ scores)        // [B,S]
{
    const int b    = blockIdx.y;
    const int n0   = blockIdx.x * NTILE;
    const int tid  = threadIdx.x;
    const int wave = tid >> 5;         // 0..7
    const int lane = tid & 31;
    const int lane16 = lane & 15;
    const int half   = lane >> 4;      // 0 or 1

    __shared__ SharedMem smem;

    smem.v[tid]    = vvec[tid];
    smem.bias[tid] = bias[b * HDIM + tid];
    if (tid < NTILE) smem.score[tid] = 0.f;

    const size_t batch_off = (size_t)b * (HDIM * (size_t)SDIM) + n0;

    v8f acc[2][4];
    const v8f vzero = {0.f, 0.f, 0.f, 0.f, 0.f, 0.f, 0.f, 0.f};
#pragma unroll
    for (int mi = 0; mi < 2; ++mi)
#pragma unroll
        for (int ni = 0; ni < 4; ++ni) acc[mi][ni] = vzero;

    // Prologue: kick off slab 0 (static rows 0..31) into buffer 0
    if (wave == 0)
        tdm_issue_slab((unsigned long long)(uintptr_t)(s_enc + batch_off), 0u);

    for (int kk = 0; kk < 16; ++kk) {
        const int k   = kk * KSTEP;   // W cols (static 0..255, dynamic 256..511)
        const int buf = kk & 1;

        __syncthreads();              // reads of buf^1 (iter kk-1) complete
        if (wave == 0) {
            if (kk < 15) {
                // issue next slab into the other buffer, then wait so that the
                // *current* slab (older, in-order) is resident: TENSORcnt <= 1
                const int kn = kk + 1;
                const float* src  = (kn < 8) ? s_enc : d_enc;
                const int    krow = (kn & 7) * KSTEP;
                tdm_issue_slab((unsigned long long)(uintptr_t)
                                   (src + batch_off + (size_t)krow * SDIM),
                               (unsigned int)((kn & 1) * SLAB_BYTES));
                __builtin_amdgcn_s_wait_tensorcnt(1);
            } else {
                __builtin_amdgcn_s_wait_tensorcnt(0);
            }
        }
        __syncthreads();              // current slab visible to all waves

        // A fragments from W (global, L2-resident): 16x32 bf16, ISA lane layout:
        //   elems 0..7  : K = k + half*8      + 0..7
        //   elems 8..15 : K = k + 16 + half*8 + 0..7
        v16bf afr[2];
#pragma unroll
        for (int mi = 0; mi < 2; ++mi) {
            const int row = wave * 32 + mi * 16 + lane16;
            const float* wp = W + (size_t)row * WLD + k + half * 8;
            v4f w0 = *(const v4f*)(wp);
            v4f w1 = *(const v4f*)(wp + 4);
            v4f w2 = *(const v4f*)(wp + 16);
            v4f w3 = *(const v4f*)(wp + 20);
            if (kk < 15) __builtin_prefetch(wp + KSTEP, 0, 3);
            v16bf a;
            a[0]  = (__bf16)w0[0]; a[1]  = (__bf16)w0[1];
            a[2]  = (__bf16)w0[2]; a[3]  = (__bf16)w0[3];
            a[4]  = (__bf16)w1[0]; a[5]  = (__bf16)w1[1];
            a[6]  = (__bf16)w1[2]; a[7]  = (__bf16)w1[3];
            a[8]  = (__bf16)w2[0]; a[9]  = (__bf16)w2[1];
            a[10] = (__bf16)w2[2]; a[11] = (__bf16)w2[3];
            a[12] = (__bf16)w3[0]; a[13] = (__bf16)w3[1];
            a[14] = (__bf16)w3[2]; a[15] = (__bf16)w3[3];
            afr[mi] = a;
        }

        // B fragments from LDS fp32 slab: N = lane16, K rows half*8+0..7 and 16+half*8+0..7
#pragma unroll
        for (int ni = 0; ni < 4; ++ni) {
            const float* bp = &smem.slab[buf][half * 8][ni * 16 + lane16];
            v4f e0 = {bp[0 * SLAB_LD], bp[1 * SLAB_LD], bp[2 * SLAB_LD], bp[3 * SLAB_LD]};
            v4f e1 = {bp[4 * SLAB_LD], bp[5 * SLAB_LD], bp[6 * SLAB_LD], bp[7 * SLAB_LD]};
            v4f e2 = {bp[16 * SLAB_LD], bp[17 * SLAB_LD], bp[18 * SLAB_LD], bp[19 * SLAB_LD]};
            v4f e3 = {bp[20 * SLAB_LD], bp[21 * SLAB_LD], bp[22 * SLAB_LD], bp[23 * SLAB_LD]};
            v16bf bf;
            bf[0]  = (__bf16)e0[0]; bf[1]  = (__bf16)e0[1];
            bf[2]  = (__bf16)e0[2]; bf[3]  = (__bf16)e0[3];
            bf[4]  = (__bf16)e1[0]; bf[5]  = (__bf16)e1[1];
            bf[6]  = (__bf16)e1[2]; bf[7]  = (__bf16)e1[3];
            bf[8]  = (__bf16)e2[0]; bf[9]  = (__bf16)e2[1];
            bf[10] = (__bf16)e2[2]; bf[11] = (__bf16)e2[3];
            bf[12] = (__bf16)e3[0]; bf[13] = (__bf16)e3[1];
            bf[14] = (__bf16)e3[2]; bf[15] = (__bf16)e3[3];
#pragma unroll
            for (int mi = 0; mi < 2; ++mi) {
                acc[mi][ni] = __builtin_amdgcn_wmma_f32_16x16x32_bf16(
                    false, afr[mi], false, bf, (short)0, acc[mi][ni], false, false);
            }
        }
    }

    // Epilogue: C layout -> element j of acc[mi][ni] is
    //   h = wave*32 + mi*16 + half*8 + j,  s = n0 + ni*16 + lane16
    float vv[16], bb[16];
#pragma unroll
    for (int idx = 0; idx < 16; ++idx) {
        const int h = wave * 32 + (idx >> 3) * 16 + half * 8 + (idx & 7);
        vv[idx] = smem.v[h];
        bb[idx] = smem.bias[h];
    }
#pragma unroll
    for (int ni = 0; ni < 4; ++ni) {
        float p = 0.f;
#pragma unroll
        for (int mi = 0; mi < 2; ++mi)
#pragma unroll
            for (int j = 0; j < 8; ++j)
                p += vv[mi * 8 + j] * tanhf(acc[mi][ni][j] + bb[mi * 8 + j]);
        atomicAdd(&smem.score[ni * 16 + lane16], p);   // ds_add_f32 cross-wave reduce
    }
    __syncthreads();
    if (tid < NTILE)
        scores[(size_t)b * SDIM + n0 + tid] = smem.score[tid];
}

// ---------------------------------------------------------------------------
// Kernel C: row softmax over S=4096, one workgroup per batch row
// ---------------------------------------------------------------------------
__global__ __launch_bounds__(256) void att_softmax_kernel(
    const float* __restrict__ scores, float* __restrict__ out)
{
    const int b = blockIdx.x;
    const int tid = threadIdx.x;
    __shared__ float red[256];
    const float* row = scores + (size_t)b * SDIM;

    float m = -3.402823466e+38f;
    for (int i = tid; i < SDIM; i += 256) m = fmaxf(m, row[i]);
    red[tid] = m;
    __syncthreads();
    for (int off = 128; off > 0; off >>= 1) {
        if (tid < off) red[tid] = fmaxf(red[tid], red[tid + off]);
        __syncthreads();
    }
    const float mx = red[0];
    __syncthreads();

    float s = 0.f;
    for (int i = tid; i < SDIM; i += 256) s += __expf(row[i] - mx);
    red[tid] = s;
    __syncthreads();
    for (int off = 128; off > 0; off >>= 1) {
        if (tid < off) red[tid] += red[tid + off];
        __syncthreads();
    }
    const float inv = 1.f / red[0];

    for (int i = tid; i < SDIM; i += 256)
        out[(size_t)b * SDIM + i] = __expf(row[i] - mx) * inv;
}

// ---------------------------------------------------------------------------
extern "C" void kernel_launch(void* const* d_in, const int* in_sizes, int n_in,
                              void* d_out, int out_size, void* d_ws, size_t ws_size,
                              hipStream_t stream) {
    const float* static_enc     = (const float*)d_in[0];   // [B,H,S]
    const float* dynamic_enc    = (const float*)d_in[1];   // [B,H,S]
    const float* decoder_hidden = (const float*)d_in[2];   // [1,B,H]
    const float* W              = (const float*)d_in[3];   // [H,3H]
    const float* v              = (const float*)d_in[4];   // [1,H]
    float* out = (float*)d_out;                            // [B,S]

    float* bias   = (float*)d_ws;              // B*H   = 16384 floats
    float* scores = bias + BATCH * HDIM;       // B*S   = 262144 floats

    att_bias_kernel<<<BATCH, 256, 0, stream>>>(W, decoder_hidden, bias);

    dim3 grid(SDIM / NTILE, BATCH);            // 64 x 64 workgroups
    att_gemm_kernel<<<grid, 256, 0, stream>>>(static_enc, dynamic_enc, W, v,
                                              bias, scores);

    att_softmax_kernel<<<BATCH, 256, 0, stream>>>(scores, out);
}
